// ProbAttention_2241972928841
// MI455X (gfx1250) — matrix operand order repaired and checked by reading, not compiled, not run on hardware
//
#include <hip/hip_runtime.h>
#include <hip/hip_bf16.h>

typedef __attribute__((ext_vector_type(16))) _Float16 v16h;
typedef __attribute__((ext_vector_type(8)))  float    v8f;
typedef __attribute__((ext_vector_type(4)))  int      v4i;

#define Bc    2
#define Lc    2048
#define Dc    512
#define Hc    8
#define DKc   64
#define SKc   40
#define NTOPc 40
#define MROWS (Bc*Lc)          // 4096

#define AS_STRIDE 40           // halfs per LDS row: 80B, 16B-aligned chunks, bank-skewed
#define PER_STAGE 6            // async 16B copies issued per thread per k-stage

// --------------------------------------------------------------------------
// CDNA5 async LDS staging (ASYNCcnt path); guarded so either toolchain builds
// --------------------------------------------------------------------------
#if defined(__has_builtin)
#if __has_builtin(__builtin_amdgcn_global_load_async_to_lds_b128)
#define HAVE_ASYNC_LDS 1
#endif
#endif

__device__ __forceinline__ void copy16(const void* g, void* l) {
#if defined(HAVE_ASYNC_LDS)
    __builtin_amdgcn_global_load_async_to_lds_b128(
        (__attribute__((address_space(1))) v4i*)(void*)g,
        (__attribute__((address_space(3))) v4i*)l, 0, 0);
#else
    *(uint4*)l = *(const uint4*)g;
#endif
}

template<int N>
__device__ __forceinline__ void wait_async() {
#if defined(HAVE_ASYNC_LDS)
#if __has_builtin(__builtin_amdgcn_s_wait_asynccnt)
    __builtin_amdgcn_s_wait_asynccnt((short)N);
#else
    asm volatile("s_wait_asynccnt %0" :: "n"(N) : "memory");
#endif
#endif
}

// ---------------------------------------------------------------------------
// elementwise fp32 -> f16
// ---------------------------------------------------------------------------
__global__ void f32_to_f16_kernel(const float* __restrict__ src,
                                  _Float16* __restrict__ dst, int n) {
    int i = blockIdx.x * blockDim.x + threadIdx.x;
    if (i < n) dst[i] = (_Float16)src[i];
}

// transpose+convert weight:  Wt[n*512+k] = (f16) W[k*512+n]
__global__ void wtransp_f16_kernel(const float* __restrict__ W,
                                   _Float16* __restrict__ Wt) {
    int tid = blockIdx.x * 256 + threadIdx.x;      // 262144 threads
    int n = tid >> 9, k = tid & (Dc - 1);
    Wt[(size_t)n * Dc + k] = (_Float16)W[(size_t)k * Dc + n];
}

// ---------------------------------------------------------------------------
// stage one BK=32 slice: A tile 128x32 f16, W^T tile 64x32 f16 -> LDS (async)
// ---------------------------------------------------------------------------
__device__ __forceinline__ void stage_tiles(const _Float16* __restrict__ Ah,
                                            const _Float16* __restrict__ Wt,
                                            _Float16* As, _Float16* Bs,
                                            int m0, int n0, int k0, int t) {
    #pragma unroll
    for (int i = 0; i < 4; ++i) {                  // 512 chunks / 128 thr
        int c = i * 128 + t;
        int row = c >> 2, j = c & 3;
        copy16(&Ah[(size_t)(m0 + row) * Dc + k0 + 8 * j],
               &As[row * AS_STRIDE + 8 * j]);
    }
    #pragma unroll
    for (int i = 0; i < 2; ++i) {                  // 256 chunks / 128 thr
        int c = i * 128 + t;
        int n = c >> 2, j = c & 3;
        copy16(&Wt[(size_t)(n0 + n) * Dc + k0 + 8 * j],
               &Bs[n * AS_STRIDE + 8 * j]);
    }
}

// ---------------------------------------------------------------------------
// WMMA GEMM:  Out[4096, 512] = Ah[4096x512] * Wt^T + bias
//   Ah f16 row-major, Wt f16 pre-transposed (Wt[n][k] = W[k][n]).
// Block 128 thr = 4 waves; tile 128(M) x 64(N); BK=32, double-buffered async.
// Wave owns 32 rows x 64 cols = 2 A frags x 4 B frags = 8 wmma / k-step.
// ---------------------------------------------------------------------------
template<bool OUT_BHLD>
__global__ __launch_bounds__(128)
void gemm_wmma_kernel(const _Float16* __restrict__ Ah,
                      const _Float16* __restrict__ Wt,
                      const float* __restrict__ bias,
                      float* __restrict__ Out)
{
    __shared__ __attribute__((aligned(16))) _Float16 As[2 * 128 * AS_STRIDE];
    __shared__ __attribute__((aligned(16))) _Float16 Bs[2 * 64 * AS_STRIDE];

    const int t     = threadIdx.x;
    const int wave  = t >> 5;
    const int lane  = t & 31;
    const int half_ = lane >> 4;
    const int l16   = lane & 15;
    const int m0    = blockIdx.x * 128;
    const int n0    = blockIdx.y * 64;

    v8f acc[2][4] = {};

    stage_tiles(Ah, Wt, As, Bs, m0, n0, 0, t);     // prologue: buffer 0

    for (int kt = 0; kt < 16; ++kt) {
        const int cur = kt & 1;
        if (kt + 1 < 16)                            // issue next stage first
            stage_tiles(Ah, Wt, As + (cur ^ 1) * 128 * AS_STRIDE,
                        Bs + (cur ^ 1) * 64 * AS_STRIDE,
                        m0, n0, (kt + 1) * 32, t);
        if (kt + 1 < 16) wait_async<PER_STAGE>();   // current buffer landed,
        else             wait_async<0>();           // next stage stays in flight
        __syncthreads();

        const _Float16* Ab = As + cur * 128 * AS_STRIDE;
        const _Float16* Bb = Bs + cur * 64 * AS_STRIDE;

        // A fragments: element i -> K = i + 8*((i>>3)+half); pairs contiguous
        union frag { v16h v; unsigned u[8]; };
        frag af[2];
        #pragma unroll
        for (int ma = 0; ma < 2; ++ma)
            #pragma unroll
            for (int j = 0; j < 8; ++j) {
                int ks = 2 * j + 8 * ((j >> 2) + half_);
                af[ma].u[j] = *(const unsigned*)
                    &Ab[(32 * wave + 16 * ma + l16) * AS_STRIDE + ks];
            }
        // B fragments: N = l16 + 16*nb, element i -> K = i + 8*((i>>3)+half)
        #pragma unroll
        for (int nb = 0; nb < 4; ++nb) {
            frag bf;
            const int coln = l16 + 16 * nb;
            #pragma unroll
            for (int j = 0; j < 8; ++j) {
                int ks = 2 * j + 8 * ((j >> 2) + half_);
                bf.u[j] = *(const unsigned*)&Bb[coln * AS_STRIDE + ks];
            }
            #pragma unroll
            for (int ma = 0; ma < 2; ++ma)
                acc[ma][nb] = __builtin_amdgcn_wmma_f32_16x16x32_f16(
                    false, af[ma].v, false, bf.v, (short)0, acc[ma][nb],
                    false, false);
        }
        __syncthreads();                            // buffer reusable
    }

    // epilogue: C/D layout -> VGPR v holds M = v + 8*half, N = l16
    #pragma unroll
    for (int ma = 0; ma < 2; ++ma)
        #pragma unroll
        for (int nb = 0; nb < 4; ++nb) {
            const int n  = n0 + 16 * nb + l16;
            const float bv = bias[n];
            #pragma unroll
            for (int v = 0; v < 8; ++v) {
                const int m   = m0 + 32 * wave + 16 * ma + v + 8 * half_;
                const float val = acc[ma][nb][v] + bv;
                if (OUT_BHLD) {
                    int b = m >> 11, l = m & (Lc - 1);
                    int h = n >> 6,  d = n & (DKc - 1);
                    Out[(((size_t)(b * Hc + h) * Lc) + l) * DKc + d] = val;
                } else {
                    Out[(size_t)m * Dc + n] = val;
                }
            }
        }
}

// ---------------------------------------------------------------------------
// Sampled sparsity measure: M[b,h,l] = max_s(q.k_idx) - sum_s(q.k_idx)/L
// ---------------------------------------------------------------------------
__global__ __launch_bounds__(256)
void sample_scores_kernel(const float* __restrict__ Qd, const float* __restrict__ Kd,
                          const int* __restrict__ idx, float* __restrict__ Mm)
{
    int tid = blockIdx.x * 256 + threadIdx.x;
    if (tid >= Bc * Hc * Lc) return;
    int l   = tid & (Lc - 1);
    int bh  = tid >> 11;
    const float* qrow = Qd + (size_t)tid * DKc;
    float qr[DKc];
    #pragma unroll
    for (int d = 0; d < DKc; ++d) qr[d] = qrow[d];

    float mx = -3.4e38f, sm = 0.f;
    for (int s = 0; s < SKc; ++s) {
        int ki = idx[l * SKc + s];
        const float* krow = Kd + ((size_t)bh * Lc + ki) * DKc;
        float acc = 0.f;
        #pragma unroll
        for (int d = 0; d < DKc; ++d) acc += qr[d] * krow[d];
        mx = fmaxf(mx, acc);
        sm += acc;
    }
    Mm[tid] = mx - sm * (1.0f / (float)Lc);
}

// ---------------------------------------------------------------------------
// Top-40 per (b,h): iterative argmax, tie -> smaller index (matches lax.top_k)
// ---------------------------------------------------------------------------
__global__ __launch_bounds__(256)
void topk_kernel(const float* __restrict__ Mm, int* __restrict__ topIdx)
{
    __shared__ float rv[256];
    __shared__ int   ri[256];
    __shared__ unsigned char sel[Lc];
    const int t  = threadIdx.x;
    const int bh = blockIdx.x;
    const float* base = Mm + (size_t)bh * Lc;

    for (int e = t; e < Lc; e += 256) sel[e] = 0;
    __syncthreads();

    for (int k = 0; k < NTOPc; ++k) {
        float best = -3.4e38f;
        int   bid  = 0x7fffffff;
        for (int e = t; e < Lc; e += 256) {
            if (!sel[e]) {
                float v = base[e];
                if (v > best || (v == best && e < bid)) { best = v; bid = e; }
            }
        }
        rv[t] = best; ri[t] = bid;
        for (int off = 128; off > 0; off >>= 1) {
            __syncthreads();
            if (t < off) {
                float v2 = rv[t + off]; int i2 = ri[t + off];
                if (v2 > rv[t] || (v2 == rv[t] && i2 < ri[t])) { rv[t] = v2; ri[t] = i2; }
            }
        }
        __syncthreads();
        if (t == 0) {
            int win = ri[0];
            topIdx[bh * NTOPc + k] = win;
            sel[win] = 1;
        }
        __syncthreads();
    }
}

// ---------------------------------------------------------------------------
// s-values: s[b,h,r] = Q[b,h,topIdx[r],:] . K[b,h,L-1,:]
// ---------------------------------------------------------------------------
__global__ __launch_bounds__(64)
void svals_kernel(const float* __restrict__ Qd, const float* __restrict__ Kd,
                  const int* __restrict__ topIdx, float* __restrict__ svals)
{
    const int bh = blockIdx.x;
    const int r  = threadIdx.x;
    if (r >= NTOPc) return;
    int c = topIdx[bh * NTOPc + r];
    const float* qrow = Qd + ((size_t)bh * Lc + c) * DKc;
    const float* krow = Kd + ((size_t)bh * Lc + (Lc - 1)) * DKc;
    float acc = 0.f;
    #pragma unroll
    for (int d = 0; d < DKc; ++d) acc += qrow[d] * krow[d];
    svals[bh * NTOPc + r] = acc;
}

// ---------------------------------------------------------------------------
// Vsum[b,h,d] = sum_k V[b,h,k,d]
// ---------------------------------------------------------------------------
__global__ __launch_bounds__(64)
void vsum_kernel(const float* __restrict__ Vd, float* __restrict__ Vsum)
{
    const int bh = blockIdx.x;
    const int d  = threadIdx.x;
    const float* v = Vd + (size_t)bh * Lc * DKc + d;
    float s = 0.f;
    for (int k = 0; k < Lc; ++k) s += v[(size_t)k * DKc];
    Vsum[bh * DKc + d] = s;
}

// ---------------------------------------------------------------------------
// Analytic softmax rows -> ctx f16 in (B*L, 512) layout:
//   r >= 40 : ctx = Vsum / L
//   r <  40 : ctx = (Vsum + (e^s - 1) V[c]) / (e^s + L - 1)
// ---------------------------------------------------------------------------
__global__ __launch_bounds__(64)
void ctx_kernel(const float* __restrict__ Vd, const float* __restrict__ Vsum,
                const int* __restrict__ topIdx, const float* __restrict__ svals,
                _Float16* __restrict__ ctxh)
{
    const int bid = blockIdx.x;
    const int b   = bid >> 14;
    const int rem = bid & 16383;
    const int h   = rem >> 11;
    const int l   = rem & (Lc - 1);
    const int d   = threadIdx.x;
    const int bh  = b * Hc + h;

    float val;
    if (l < NTOPc) {
        int   c = topIdx[bh * NTOPc + l];
        float s = svals[bh * NTOPc + l];
        float w = __expf(s);
        float z = w + (float)(Lc - 1);
        float vc = Vd[((size_t)bh * Lc + c) * DKc + d];
        val = (Vsum[bh * DKc + d] + (w - 1.0f) * vc) / z;
    } else {
        val = Vsum[bh * DKc + d] * (1.0f / (float)Lc);
    }
    ctxh[((size_t)(b * Lc + l)) * Dc + h * DKc + d] = (_Float16)val;
}

// ---------------------------------------------------------------------------
// launch
// ---------------------------------------------------------------------------
extern "C" void kernel_launch(void* const* d_in, const int* in_sizes, int n_in,
                              void* d_out, int out_size, void* d_ws, size_t ws_size,
                              hipStream_t stream)
{
    const float* q  = (const float*)d_in[0];
    const float* k  = (const float*)d_in[1];
    const float* v  = (const float*)d_in[2];
    const float* Wq = (const float*)d_in[3];
    const float* bq = (const float*)d_in[4];
    const float* Wk = (const float*)d_in[5];
    const float* bk = (const float*)d_in[6];
    const float* Wv = (const float*)d_in[7];
    const float* bv = (const float*)d_in[8];
    const float* Wo = (const float*)d_in[9];
    const float* bo = (const float*)d_in[10];
    const int* idx  = (const int*)d_in[11];
    float* out = (float*)d_out;

    char* ws = (char*)d_ws;
    size_t off = 0;
    auto alloc = [&](size_t bytes) -> char* {
        char* p = ws + off;
        off += (bytes + 255) & ~(size_t)255;
        return p;
    };
    const size_t WN = (size_t)Dc * Dc;                       // 262144
    const size_t AN = (size_t)MROWS * Dc;                    // 2097152
    _Float16* hWqT = (_Float16*)alloc(WN * 2);
    _Float16* hWkT = (_Float16*)alloc(WN * 2);
    _Float16* hWvT = (_Float16*)alloc(WN * 2);
    _Float16* hWoT = (_Float16*)alloc(WN * 2);
    _Float16* hq   = (_Float16*)alloc(AN * 2);
    _Float16* hk   = (_Float16*)alloc(AN * 2);
    _Float16* hv   = (_Float16*)alloc(AN * 2);
    float* Qd     = (float*)alloc(AN * 4);                   // (B,H,L,64)
    float* Kd     = (float*)alloc(AN * 4);
    float* Vd     = (float*)alloc(AN * 4);
    float* Mm     = (float*)alloc((size_t)Bc * Hc * Lc * 4);
    int*   topIdx = (int*)alloc((size_t)Bc * Hc * NTOPc * 4);
    float* svals  = (float*)alloc((size_t)Bc * Hc * NTOPc * 4);
    float* Vsum   = (float*)alloc((size_t)Bc * Hc * DKc * 4);
    _Float16* ctxh = (_Float16*)alloc(AN * 2);

    // preprocess: weights -> f16 transposed; activations -> f16
    int wb = ((int)WN + 255) / 256;
    wtransp_f16_kernel<<<wb, 256, 0, stream>>>(Wq, hWqT);
    wtransp_f16_kernel<<<wb, 256, 0, stream>>>(Wk, hWkT);
    wtransp_f16_kernel<<<wb, 256, 0, stream>>>(Wv, hWvT);
    wtransp_f16_kernel<<<wb, 256, 0, stream>>>(Wo, hWoT);
    int ab = ((int)AN + 255) / 256;
    f32_to_f16_kernel<<<ab, 256, 0, stream>>>(q, hq, (int)AN);
    f32_to_f16_kernel<<<ab, 256, 0, stream>>>(k, hk, (int)AN);
    f32_to_f16_kernel<<<ab, 256, 0, stream>>>(v, hv, (int)AN);

    // projections, output in (B,H,L,64)
    dim3 ggrid(MROWS / 128, Dc / 64);
    gemm_wmma_kernel<true><<<ggrid, 128, 0, stream>>>(hq, hWqT, bq, Qd);
    gemm_wmma_kernel<true><<<ggrid, 128, 0, stream>>>(hk, hWkT, bk, Kd);
    gemm_wmma_kernel<true><<<ggrid, 128, 0, stream>>>(hv, hWvT, bv, Vd);

    // sparsity measure, top-k, s-values, Vsum, analytic-softmax context
    sample_scores_kernel<<<(Bc * Hc * Lc) / 256, 256, 0, stream>>>(Qd, Kd, idx, Mm);
    topk_kernel<<<Bc * Hc, 256, 0, stream>>>(Mm, topIdx);
    svals_kernel<<<Bc * Hc, 64, 0, stream>>>(Qd, Kd, topIdx, svals);
    vsum_kernel<<<Bc * Hc, 64, 0, stream>>>(Vd, Vsum);
    ctx_kernel<<<Bc * Hc * Lc, 64, 0, stream>>>(Vd, Vsum, topIdx, svals, ctxh);

    // output projection, row-major output
    gemm_wmma_kernel<false><<<ggrid, 128, 0, stream>>>(ctxh, hWoT, bo, out);
}
